// MultiHeadSelfAttention_14963666059606
// MI455X (gfx1250) — matrix operand order repaired and checked by reading, not compiled, not run on hardware
//
#include <hip/hip_runtime.h>
#include <hip/hip_bf16.h>

// MI455X / gfx1250 MHSA forward.
// Compute-bound (~137 GFLOP vs ~0.2 GB HBM => HBM time ~8us << WMMA time), so:
//  - all matmuls on v_wmma_f32_16x16x32_f16 (f16 in, f32 accum)
//  - 64x64 per-wave register tiles (16 wmma per 8 LDS fragment loads)
//  - f32->f16 conversion/transpose hoisted into one-shot pre-kernels so the
//    hot loops stage tiles with async global->LDS DMA (ASYNCcnt), double
//    buffered to overlap memory with matrix math.

#ifndef USE_ASYNC_LDS
#define USE_ASYNC_LDS 1
#endif

typedef __attribute__((ext_vector_type(16))) _Float16 v16h;
typedef __attribute__((ext_vector_type(8)))  _Float16 v8h;
typedef __attribute__((ext_vector_type(8)))  float    v8f;

#define EMB   1024
#define HEADS 16
#define HDIM  64
#define BATCH 4
#define SEQ   2048
#define NTOK  (BATCH*SEQ)   // 8192

static __device__ __forceinline__ v16h cat8(v8h lo, v8h hi) {
  return __builtin_shufflevector(lo, hi, 0,1,2,3,4,5,6,7,8,9,10,11,12,13,14,15);
}

static __device__ __forceinline__ v8f wmma_f16(v16h a, v16h b, v8f c) {
  // (neg_a, A, neg_b, B, c_mod, C, reuse_a, reuse_b)
  return __builtin_amdgcn_wmma_f32_16x16x32_f16(false, a, false, b, (short)0, c,
                                                false, false);
}

// 16B global -> LDS copy. Async path: CDNA5 async DMA (tracked by ASYNCcnt).
static __device__ __forceinline__ void copy16(const _Float16* __restrict__ g,
                                              _Float16* l) {
#if USE_ASYNC_LDS
  unsigned loff = (unsigned)(size_t)(__attribute__((address_space(3))) _Float16*)l;
  asm volatile("global_load_async_to_lds_b128 %0, %1, off"
               :: "v"(loff), "v"(g) : "memory");
#else
  *(v8h*)l = *(const v8h*)g;
#endif
}

template<int N> static __device__ __forceinline__ void wait_async_le() {
#if USE_ASYNC_LDS
  asm volatile("s_wait_asynccnt %0" :: "i"(N) : "memory");
#endif
}

// ---------------------------------------------------------------------------
// One-shot pre-kernels (bandwidth-trivial): f32 -> f16 convert, and
// convert + transpose for weights so GEMM B-tiles are contiguous along K.
// ---------------------------------------------------------------------------
__global__ __launch_bounds__(256)
void convert_f16(const float* __restrict__ in, _Float16* __restrict__ out) {
  const size_t i = ((size_t)blockIdx.x * 256 + threadIdx.x) * 4;
  const float4 v = *(const float4*)(in + i);
  _Float16* o = out + i;
  o[0] = (_Float16)v.x; o[1] = (_Float16)v.y;
  o[2] = (_Float16)v.z; o[3] = (_Float16)v.w;
}

__global__ __launch_bounds__(256)
void transpose_f16(const float* __restrict__ in, _Float16* __restrict__ out,
                   int K, int N) {   // in[K][N] f32 -> out[N][K] f16
  __shared__ _Float16 t[32][33];
  const int n0 = blockIdx.x * 32, k0 = blockIdx.y * 32;
  const int tx = threadIdx.x & 31, ty = threadIdx.x >> 5;
  #pragma unroll
  for (int i = 0; i < 4; ++i) {
    const int k = ty + i * 8;
    t[k][tx] = (_Float16)in[(size_t)(k0 + k) * N + n0 + tx];
  }
  __syncthreads();
  #pragma unroll
  for (int i = 0; i < 4; ++i) {
    const int n = ty + i * 8;
    out[(size_t)(n0 + n) * K + k0 + tx] = t[tx][n];
  }
}

// ---------------------------------------------------------------------------
// GEMM: C[M,N] = A[M,K] * Bt[N,K]^T + bias. All-f16 operands.
// Block tile 128x256, 8 waves (2M x 4N), wave tile 64x64 (16 wmma / K-step).
// Double-buffered async LDS staging.
// ---------------------------------------------------------------------------
template<bool QKV_STORE>
__global__ __launch_bounds__(256)
void gemm_f16(const _Float16* __restrict__ A, const _Float16* __restrict__ Bt,
              const float* __restrict__ bias, float* __restrict__ Cout,
              _Float16* __restrict__ Qh, _Float16* __restrict__ Kh,
              _Float16* __restrict__ Vt, int Ktot, int Ntot)
{
  __shared__ _Float16 AsBuf[2 * 128 * 40];   // [row][k] pad to 40 halves
  __shared__ _Float16 BsBuf[2 * 256 * 40];   // [n][k]

  const int tid  = threadIdx.x;
  const int lane = tid & 31;
  const int w    = tid >> 5;
  const int wm   = w & 1, wn = w >> 1;       // 2 x 4 wave grid
  const int hl   = lane >> 4, lr = lane & 15;
  const int blockN = blockIdx.x * 256;
  const int blockM = blockIdx.y * 128;

  v8f acc[4][4] = {};

  auto stage = [&](int kt, int buf) {        // 6 async ops per thread
    const int k0 = kt << 5;
    _Float16* As = &AsBuf[buf * (128 * 40)];
    _Float16* Bs = &BsBuf[buf * (256 * 40)];
    #pragma unroll
    for (int i = 0; i < 2; ++i) {            // A tile: 128 x 32 halves
      const int cid = i * 256 + tid, row = cid >> 2, c = cid & 3;
      copy16(A + (size_t)(blockM + row) * Ktot + k0 + c * 8, &As[row * 40 + c * 8]);
    }
    #pragma unroll
    for (int i = 0; i < 4; ++i) {            // B tile: 256 x 32 halves
      const int cid = i * 256 + tid, nrow = cid >> 2, c = cid & 3;
      copy16(Bt + (size_t)(blockN + nrow) * Ktot + k0 + c * 8, &Bs[nrow * 40 + c * 8]);
    }
  };

  const int ksteps = Ktot >> 5;
  stage(0, 0);
  for (int kt = 0; kt < ksteps; ++kt) {
    if (kt + 1 < ksteps) { stage(kt + 1, (kt + 1) & 1); wait_async_le<6>(); }
    else                 { wait_async_le<0>(); }
    __syncthreads();

    const _Float16* As = &AsBuf[(kt & 1) * (128 * 40)];
    const _Float16* Bs = &BsBuf[(kt & 1) * (256 * 40)];
    v16h af[4], bf[4];
    #pragma unroll
    for (int mi = 0; mi < 4; ++mi) {
      const _Float16* pr = &As[(wm * 64 + mi * 16 + lr) * 40];
      // A 16x32 frag: lane = M; chunks at K = 8*hl and 16+8*hl
      af[mi] = cat8(*(const v8h*)(pr + hl * 8), *(const v8h*)(pr + 16 + hl * 8));
    }
    #pragma unroll
    for (int ni = 0; ni < 4; ++ni) {
      const _Float16* pr = &Bs[(wn * 64 + ni * 16 + lr) * 40];
      // B 32x16 frag: lane = N; chunks at K = 8*hl and 16+8*hl
      bf[ni] = cat8(*(const v8h*)(pr + hl * 8), *(const v8h*)(pr + 16 + hl * 8));
    }
    #pragma unroll
    for (int mi = 0; mi < 4; ++mi)
      #pragma unroll
      for (int ni = 0; ni < 4; ++ni)
        acc[mi][ni] = wmma_f16(af[mi], bf[ni], acc[mi][ni]);
    __syncthreads();
  }

  // epilogue: bias + store. C layout: VGPR r -> M = r + 8*hl, N = lane%16
  #pragma unroll
  for (int ni = 0; ni < 4; ++ni) {
    const int n  = blockN + wn * 64 + ni * 16 + lr;
    const float bv = bias[n];
    #pragma unroll
    for (int mi = 0; mi < 4; ++mi) {
      #pragma unroll
      for (int r = 0; r < 8; ++r) {
        const int m = blockM + wm * 64 + mi * 16 + r + hl * 8;
        const float val = acc[mi][ni][r] + bv;
        if (QKV_STORE) {
          const int s = n >> 10, rem = n & 1023;
          const int h = rem >> 6, dh = rem & 63;
          const int b = m >> 11, t = m & (SEQ - 1);
          const int bh = (b << 4) | h;
          if (s == 0)      Qh[((size_t)bh * SEQ + t) * HDIM + dh] = (_Float16)val;
          else if (s == 1) Kh[((size_t)bh * SEQ + t) * HDIM + dh] = (_Float16)val;
          else             Vt[((size_t)bh * HDIM + dh) * SEQ + t] = (_Float16)val;
        } else {
          Cout[(size_t)m * Ntot + n] = val;
        }
      }
    }
  }
}

// ---------------------------------------------------------------------------
// Flash attention: block = (b,h, 128 q-rows), 8 waves x 16 q-rows.
// 64-key chunks staged via double-buffered async LDS copies.
// ---------------------------------------------------------------------------
__global__ __launch_bounds__(256)
void attn_kernel(const _Float16* __restrict__ Qh, const _Float16* __restrict__ Kh,
                 const _Float16* __restrict__ Vt, _Float16* __restrict__ Oh)
{
  __shared__ _Float16 KsBuf[2 * 64 * 72];   // [key][d], stride 72 halves
  __shared__ _Float16 VsBuf[2 * 64 * 72];   // V^T: [d][key]
  __shared__ _Float16 Pb[8 * 16 * 72];      // per-wave P staging [m][key]

  const int tid  = threadIdx.x;
  const int lane = tid & 31;
  const int w    = tid >> 5;
  const int hl   = lane >> 4, lr = lane & 15;
  const int qt   = blockIdx.x;              // 0..15 q tiles of 128
  const int bh   = blockIdx.y;              // 0..63
  const int b    = bh >> 4, h = bh & 15;

  const size_t kvbase = (size_t)bh * SEQ * HDIM;

  // Q fragments (2 frags across d=0..63), scale 1/8 folded in once.
  const int qrow = qt * 128 + w * 16 + lr;
  const _Float16* qp = Qh + kvbase + (size_t)qrow * HDIM;
  v16h qa0 = cat8(*(const v8h*)(qp + hl * 8),      *(const v8h*)(qp + 16 + hl * 8));
  v16h qa1 = cat8(*(const v8h*)(qp + 32 + hl * 8), *(const v8h*)(qp + 48 + hl * 8));
  #pragma unroll
  for (int j = 0; j < 16; ++j) {
    qa0[j] *= (_Float16)0.125f;
    qa1[j] *= (_Float16)0.125f;
  }

  float mrow[8], lsum[8];
  v8f oacc[4] = {};
  #pragma unroll
  for (int r = 0; r < 8; ++r) { mrow[r] = -1e30f; lsum[r] = 0.0f; }

  _Float16* Pw = &Pb[w * 16 * 72];

  auto stage = [&](int kt, int buf) {       // 4 async ops per thread
    _Float16* Ks = &KsBuf[buf * (64 * 72)];
    _Float16* Vs = &VsBuf[buf * (64 * 72)];
    #pragma unroll
    for (int i = 0; i < 2; ++i) {
      const int cid = i * 256 + tid;        // 0..511 chunks of 8 halves
      { const int key = cid >> 3, c8 = cid & 7;
        copy16(Kh + kvbase + (size_t)(kt * 64 + key) * HDIM + c8 * 8,
               &Ks[key * 72 + c8 * 8]); }
      { const int d = cid >> 3, c8 = cid & 7;
        copy16(Vt + kvbase + (size_t)d * SEQ + kt * 64 + c8 * 8,
               &Vs[d * 72 + c8 * 8]); }
    }
  };

  const int steps = SEQ / 64;               // 32
  stage(0, 0);
  for (int kt = 0; kt < steps; ++kt) {
    if (kt + 1 < steps) { stage(kt + 1, (kt + 1) & 1); wait_async_le<4>(); }
    else                { wait_async_le<0>(); }
    __syncthreads();

    const _Float16* Ks = &KsBuf[(kt & 1) * (64 * 72)];
    const _Float16* Vs = &VsBuf[(kt & 1) * (64 * 72)];

    // ---- S = (Q/8) K^T for four 16-key sub-tiles ----
    v8f s[4];
    #pragma unroll
    for (int kk = 0; kk < 4; ++kk) {
      const _Float16* kp = &Ks[(kk * 16 + lr) * 72];  // lane = N = key
      v16h kb0 = cat8(*(const v8h*)(kp + hl * 8),      *(const v8h*)(kp + 16 + hl * 8));
      v16h kb1 = cat8(*(const v8h*)(kp + 32 + hl * 8), *(const v8h*)(kp + 48 + hl * 8));
      v8f z = {};
      z = wmma_f16(qa0, kb0, z);
      z = wmma_f16(qa1, kb1, z);
      s[kk] = z;
    }

    // ---- online softmax; row m = r + 8*hl lives across a 16-lane group ----
    #pragma unroll
    for (int r = 0; r < 8; ++r) {
      float mx = fmaxf(fmaxf(s[0][r], s[1][r]), fmaxf(s[2][r], s[3][r]));
      #pragma unroll
      for (int off = 8; off >= 1; off >>= 1) mx = fmaxf(mx, __shfl_xor(mx, off, 16));
      const float mnew  = fmaxf(mrow[r], mx);
      const float alpha = __expf(mrow[r] - mnew);
      mrow[r] = mnew;
      const float p0 = __expf(s[0][r] - mnew);
      const float p1 = __expf(s[1][r] - mnew);
      const float p2 = __expf(s[2][r] - mnew);
      const float p3 = __expf(s[3][r] - mnew);
      float ps = (p0 + p1) + (p2 + p3);
      #pragma unroll
      for (int off = 8; off >= 1; off >>= 1) ps += __shfl_xor(ps, off, 16);
      lsum[r] = lsum[r] * alpha + ps;
      #pragma unroll
      for (int ni = 0; ni < 4; ++ni) oacc[ni][r] *= alpha;
      _Float16* pr = &Pw[(r + hl * 8) * 72];
      pr[lr]      = (_Float16)p0;
      pr[16 + lr] = (_Float16)p1;
      pr[32 + lr] = (_Float16)p2;
      pr[48 + lr] = (_Float16)p3;
    }
    // same-wave LDS ops are in-order; keep the compiler honest too
    asm volatile("s_wait_dscnt 0" ::: "memory");

    // ---- O += P (16x64) x V (64x64) ----
    const _Float16* pp = &Pw[lr * 72];
    v16h pa0 = cat8(*(const v8h*)(pp + hl * 8),      *(const v8h*)(pp + 16 + hl * 8));
    v16h pa1 = cat8(*(const v8h*)(pp + 32 + hl * 8), *(const v8h*)(pp + 48 + hl * 8));
    #pragma unroll
    for (int ni = 0; ni < 4; ++ni) {
      const _Float16* vp = &Vs[(ni * 16 + lr) * 72];  // lane = N = d
      v16h vb0 = cat8(*(const v8h*)(vp + hl * 8),      *(const v8h*)(vp + 16 + hl * 8));
      v16h vb1 = cat8(*(const v8h*)(vp + 32 + hl * 8), *(const v8h*)(vp + 48 + hl * 8));
      oacc[ni] = wmma_f16(pa0, vb0, oacc[ni]);
      oacc[ni] = wmma_f16(pa1, vb1, oacc[ni]);
    }
    __syncthreads();
  }

  // ---- normalize and store O as f16 in [B*T, E] layout for GEMM2 ----
  #pragma unroll
  for (int ni = 0; ni < 4; ++ni) {
    #pragma unroll
    for (int r = 0; r < 8; ++r) {
      const int m = qt * 128 + w * 16 + r + hl * 8;
      const int d = ni * 16 + lr;
      const float val = oacc[ni][r] / lsum[r];
      Oh[((size_t)(b * SEQ + m) * HEADS + h) * HDIM + d] = (_Float16)val;
    }
  }
}

extern "C" void kernel_launch(void* const* d_in, const int* in_sizes, int n_in,
                              void* d_out, int out_size, void* d_ws, size_t ws_size,
                              hipStream_t stream)
{
  (void)in_sizes; (void)n_in; (void)out_size; (void)ws_size;
  const float* x     = (const float*)d_in[0];
  const float* qkv_w = (const float*)d_in[1];
  const float* qkv_b = (const float*)d_in[2];
  const float* fc_w  = (const float*)d_in[3];
  const float* fc_b  = (const float*)d_in[4];
  float* out = (float*)d_out;

  char* ws = (char*)d_ws;                         // 88 MB carved
  _Float16* Qh  = (_Float16*)(ws + ((size_t) 0 << 20));
  _Float16* Kh  = (_Float16*)(ws + ((size_t)16 << 20));
  _Float16* Vt  = (_Float16*)(ws + ((size_t)32 << 20));
  _Float16* Oh  = (_Float16*)(ws + ((size_t)48 << 20));
  _Float16* Xh  = (_Float16*)(ws + ((size_t)64 << 20));
  _Float16* Wqt = (_Float16*)(ws + ((size_t)80 << 20));
  _Float16* Wft = (_Float16*)(ws + ((size_t)86 << 20));

  // 0) one-shot convert / transpose (bandwidth-trivial)
  convert_f16<<<(NTOK * EMB) / (256 * 4), 256, 0, stream>>>(x, Xh);
  transpose_f16<<<dim3(3 * EMB / 32, EMB / 32), 256, 0, stream>>>(qkv_w, Wqt, EMB, 3 * EMB);
  transpose_f16<<<dim3(EMB / 32, EMB / 32), 256, 0, stream>>>(fc_w, Wft, EMB, EMB);

  // 1) QKV projection [8192,1024] x [1024,3072] + b -> f16 Q, K, V^T
  gemm_f16<true><<<dim3(3 * EMB / 256, NTOK / 128), 256, 0, stream>>>(
      Xh, Wqt, qkv_b, nullptr, Qh, Kh, Vt, EMB, 3 * EMB);

  // 2) flash attention per (b,h), 128 q-rows per block
  attn_kernel<<<dim3(SEQ / 128, BATCH * HEADS), 256, 0, stream>>>(Qh, Kh, Vt, Oh);

  // 3) output projection [8192,1024] x [1024,1024] + b -> f32 out
  gemm_f16<false><<<dim3(EMB / 256, NTOK / 128), 256, 0, stream>>>(
      Oh, Wft, fc_b, out, nullptr, nullptr, nullptr, EMB, EMB);
}
